// Block_80032420594011
// MI455X (gfx1250) — matrix operand order repaired and checked by reading, not compiled, not run on hardware
//
#include <hip/hip_runtime.h>
#include <hip/hip_bf16.h>
#include <math.h>

// ---------------------------------------------------------------------------
// CDNA5 (gfx1250) transformer block: bf16 WMMA GEMMs (TDM-fed, double-
// buffered LDS) + flash attention. Wave32, V_WMMA_F32_16X16X32_BF16.
// ---------------------------------------------------------------------------

typedef __attribute__((ext_vector_type(16))) __bf16 v16bf;
typedef __attribute__((ext_vector_type(8)))  float  v8f;
typedef __attribute__((ext_vector_type(8)))  short  short8;
typedef __attribute__((ext_vector_type(16))) short  short16;
typedef __attribute__((ext_vector_type(4)))  unsigned short ushort4v;
typedef __attribute__((ext_vector_type(4)))  unsigned int   uint4v;
typedef __attribute__((ext_vector_type(8)))  unsigned int   uint8v;

union FragAB { short16 v; short8 h[2]; };

__device__ __forceinline__ unsigned short f2bf(float f) {
    unsigned int u = __float_as_uint(f);
    u += 0x7fffu + ((u >> 16) & 1u);          // round-to-nearest-even
    return (unsigned short)(u >> 16);
}

__device__ __forceinline__ v8f vzero8() {
    v8f z = {0.f, 0.f, 0.f, 0.f, 0.f, 0.f, 0.f, 0.f};
    return z;
}

__device__ __forceinline__ v8f wmma_bf16(const FragAB& a, const FragAB& b, v8f c) {
    return __builtin_amdgcn_wmma_f32_16x16x32_bf16(
        false, __builtin_bit_cast(v16bf, a.v),
        false, __builtin_bit_cast(v16bf, b.v),
        (short)0, c, false, false);
}

// ---------------------------------------------------------------------------
// Tensor Data Mover: 2-D bf16 tile (tile_w x tile_h elements) from global to
// LDS, with hardware LDS padding of 4 DWORDs (8 halves) after every 16 DWORDs
// (32 halves) -> LDS row stride of 40 halves, matching the fragment readers.
// D# per CDNA5 ISA 8.3/8.4; groups 2/3 omitted (2-D tensor).
// ---------------------------------------------------------------------------
__device__ __forceinline__ void tdm_load_tile_2d(unsigned lds_byte,
                                                 const unsigned short* gptr,
                                                 unsigned tile_w, unsigned tile_h,
                                                 unsigned row_stride_elems)
{
    const unsigned long long ga = (unsigned long long)(uintptr_t)gptr;
    uint4v g0;
    g0[0] = 1u;                                    // count=1 valid user descriptor
    g0[1] = lds_byte;                              // lds_addr (bytes)
    g0[2] = (unsigned)ga;                          // global_addr[31:0]
    g0[3] = (unsigned)((ga >> 32) & 0x01ffffffu)   // global_addr[56:32]
          | 0x80000000u;                           // type=2 ("image")
    uint8v g1;
    g1[0] = (1u << 16)                             // data_size: 2 bytes
          | (1u << 20)                             // pad_enable
          | (3u << 22)                             // pad_interval: 16 DWORDs (64B row)
          | (3u << 25);                            // pad_amount: 4 DWORDs (16B pad)
    g1[1] = (tile_w & 0xffffu) << 16;                            // tensor_dim0 lo
    g1[2] = (tile_w >> 16) | ((tile_h & 0xffffu) << 16);         // dim0 hi | dim1 lo
    g1[3] = (tile_h >> 16) | ((tile_w & 0xffffu) << 16);         // dim1 hi | tile_dim0
    g1[4] = tile_h & 0xffffu;                                    // tile_dim1 (tile_dim2=0)
    g1[5] = row_stride_elems;                                    // tensor_dim0_stride lo32
    g1[6] = 0u;                                                  // stride0 hi | dim1_stride lo
    g1[7] = 0u;
    asm volatile("tensor_load_to_lds %0, %1" :: "s"(g0), "s"(g1) : "memory");
}

// ---------------------------------------------------------------------------
// fp32 -> bf16 straight conversion (x input), 4 elements / thread
// ---------------------------------------------------------------------------
__global__ __launch_bounds__(256)
void cvt_bf16_kernel(const float* __restrict__ s, unsigned short* __restrict__ d, long n4) {
    const long i = (long)blockIdx.x * 256 + threadIdx.x;
    if (i >= n4) return;
    const float4 v = ((const float4*)s)[i];
    ushort4v o = { f2bf(v.x), f2bf(v.y), f2bf(v.z), f2bf(v.w) };
    ((ushort4v*)d)[i] = o;
}

// ---------------------------------------------------------------------------
// fp32 [K,N] -> bf16 [N,K] (weights pre-transposed so GEMM B-fragments are
// contiguous-K loads, matching the CDNA5 WMMA B-matrix lane layout)
// ---------------------------------------------------------------------------
__global__ __launch_bounds__(256)
void cvt_transpose_kernel(const float* __restrict__ src, unsigned short* __restrict__ dst,
                          int K, int N) {
    __shared__ float t[32][33];
    const int bx = blockIdx.x * 32;     // N
    const int by = blockIdx.y * 32;     // K
    const int lx = threadIdx.x & 31;
    const int ly = threadIdx.x >> 5;    // 0..7
#pragma unroll
    for (int j = 0; j < 32; j += 8)
        t[ly + j][lx] = src[(long)(by + ly + j) * N + bx + lx];
    __syncthreads();
#pragma unroll
    for (int j = 0; j < 32; j += 8)
        dst[(long)(bx + ly + j) * K + by + lx] = f2bf(t[lx][ly + j]);
}

// ---------------------------------------------------------------------------
// GEMM: out = epilogue(A[M,K](bf16) @ Wt[N,K]^T(bf16) + bias [+gelu] [+resid])
// 128x128 block tile, 8 waves (4Mx2N), each wave 32x64 (2x4 16x16 tiles).
// K chunked by 32; tiles streamed into double-buffered LDS by the TDM
// (wave 0 posts descriptors; TENSORcnt + one barrier per chunk).
// ---------------------------------------------------------------------------
__global__ __launch_bounds__(256)
void gemm_bf16_wmma(const unsigned short* __restrict__ A,
                    const unsigned short* __restrict__ Wt,
                    const float* __restrict__ bias,
                    const float* __restrict__ resid,
                    float* __restrict__ outF,
                    unsigned short* __restrict__ outB,
                    int M, int N, int K, int gelu)
{
    constexpr int LDT = 40;                       // padded LDS stride (halves)
    __shared__ unsigned short As[2][128 * LDT];
    __shared__ unsigned short Bs[2][128 * LDT];

    const int tid  = threadIdx.x;
    const int lane = tid & 31;
    const int w    = tid >> 5;
    const int wm   = w & 3;
    const int wn   = w >> 2;
    const int lrow = lane & 15;
    const int hi   = lane >> 4;
    const long blockM = (long)blockIdx.y * 128;
    const long blockN = (long)blockIdx.x * 128;

    v8f acc[2][4];
#pragma unroll
    for (int mt = 0; mt < 2; ++mt)
#pragma unroll
        for (int nt = 0; nt < 4; ++nt) acc[mt][nt] = vzero8();

    const int nkc = K >> 5;

    auto issue = [&](int buf, int k0) {
        tdm_load_tile_2d((unsigned)(uintptr_t)&As[buf][0],
                         A + blockM * (long)K + k0, 32u, 128u, (unsigned)K);
        tdm_load_tile_2d((unsigned)(uintptr_t)&Bs[buf][0],
                         Wt + blockN * (long)K + k0, 32u, 128u, (unsigned)K);
    };

    if (w == 0) {
        issue(0, 0);
        __builtin_amdgcn_s_wait_tensorcnt(0);
    }
    __syncthreads();

    for (int kc = 0; kc < nkc; ++kc) {
        const int cur = kc & 1;
        if (w == 0 && kc + 1 < nkc) issue(cur ^ 1, (kc + 1) << 5);   // overlap DMA

        const unsigned short* Ab = &As[cur][0];
        const unsigned short* Bb = &Bs[cur][0];
        FragAB af[2], bf[4];
#pragma unroll
        for (int mt = 0; mt < 2; ++mt) {          // A: rows on lanes, K halves 0..7 / 16..23
            const unsigned short* p = Ab + (wm * 32 + mt * 16 + lrow) * LDT + hi * 8;
            af[mt].h[0] = *(const short8*)p;
            af[mt].h[1] = *(const short8*)(p + 16);
        }
#pragma unroll
        for (int nt = 0; nt < 4; ++nt) {          // B: cols on lanes, K halves 0..7 / 16..23
            const unsigned short* p = Bb + (wn * 64 + nt * 16 + lrow) * LDT + hi * 8;
            bf[nt].h[0] = *(const short8*)p;
            bf[nt].h[1] = *(const short8*)(p + 16);
        }
#pragma unroll
        for (int mt = 0; mt < 2; ++mt)
#pragma unroll
            for (int nt = 0; nt < 4; ++nt)
                acc[mt][nt] = wmma_bf16(af[mt], bf[nt], acc[mt][nt]);

        if (w == 0) __builtin_amdgcn_s_wait_tensorcnt(0);
        __syncthreads();
    }

    // Epilogue: C layout = VGPR i -> row (i + hi*8), col = lrow
#pragma unroll
    for (int mt = 0; mt < 2; ++mt) {
#pragma unroll
        for (int nt = 0; nt < 4; ++nt) {
            const long gm = blockM + wm * 32 + mt * 16 + hi * 8;
            const long gn = blockN + wn * 64 + nt * 16 + lrow;
            const float bv = bias ? bias[gn] : 0.0f;
#pragma unroll
            for (int i = 0; i < 8; ++i) {
                float v = acc[mt][nt][i] + bv;
                if (gelu) v = 0.5f * v * (1.0f + erff(v * 0.70710678f));
                const long off = (gm + i) * (long)N + gn;
                if (resid) v += resid[off];
                if (outF) outF[off] = v;
                if (outB) outB[off] = f2bf(v);
            }
        }
    }
}

// ---------------------------------------------------------------------------
// Causal flash attention: block = 128 q rows (8 waves x 16), 64-key tiles,
// D = 128. S = Q K^T via WMMA, online softmax, O += P V via WMMA.
// ---------------------------------------------------------------------------
__global__ __launch_bounds__(256)
void attention_wmma(const unsigned short* __restrict__ qkv,   // [B*T, 3C] bf16
                    unsigned short* __restrict__ outb,        // [B*T, C]  bf16
                    int T, int C, int nhead)
{
    constexpr int KL = 136;   // K-tile stride (128 + 8 halves)
    constexpr int VL = 72;    // Vt / P stride (64 + 8 halves)
    __shared__ unsigned short Ks[64 * KL];        // K tile [key][d]
    __shared__ unsigned short Vs[128 * VL];       // Vt tile [d][key]
    __shared__ unsigned short Ps[8 * 16 * VL];    // per-wave P staging [qrow][key]

    const int D    = C / nhead;                   // 128
    const int R3   = 3 * C;
    const int tid  = threadIdx.x;
    const int lane = tid & 31;
    const int w    = tid >> 5;
    const int lrow = lane & 15;
    const int hi   = lane >> 4;
    const int qt   = blockIdx.x;
    const int h    = blockIdx.y;
    const int b    = blockIdx.z;
    const long rowBase = (long)b * T;
    const int qrow0 = qt * 128 + w * 16;

    // Q A-fragments direct from global: per-lane d = hi*8+0..7 and 16+hi*8+0..7 of each 32-chunk (A layout)
    FragAB qf[4];
    {
        const unsigned short* qp = qkv + (rowBase + qrow0 + lrow) * (long)R3 + h * D;
#pragma unroll
        for (int c = 0; c < 4; ++c) {
            const unsigned short* p = qp + c * 32 + hi * 8;
            qf[c].h[0] = *(const short8*)p;
            qf[c].h[1] = *(const short8*)(p + 16);
        }
    }

    v8f o[8];
#pragma unroll
    for (int t = 0; t < 8; ++t) o[t] = vzero8();
    float rm[8], rl[8];
#pragma unroll
    for (int i = 0; i < 8; ++i) { rm[i] = -3.0e38f; rl[i] = 0.0f; }

    const float scale = 0.088388347648318447f;    // 1/sqrt(128)
    const int kend = (qt + 1) * 128;

    for (int kb = 0; kb < kend; kb += 64) {
        {   // cooperative load: K row-major, V transposed
            const int key  = tid >> 2;            // 0..63
            const int dseg = (tid & 3) * 32;
            const unsigned short* kp = qkv + (rowBase + kb + key) * (long)R3 + C + h * D + dseg;
            const short8 k0 = *(const short8*)(kp);
            const short8 k1 = *(const short8*)(kp + 8);
            const short8 k2 = *(const short8*)(kp + 16);
            const short8 k3 = *(const short8*)(kp + 24);
            const unsigned short* vp = qkv + (rowBase + kb + key) * (long)R3 + 2 * C + h * D + dseg;
            union { short8 s[4]; unsigned short u[32]; } vb;
            vb.s[0] = *(const short8*)(vp);
            vb.s[1] = *(const short8*)(vp + 8);
            vb.s[2] = *(const short8*)(vp + 16);
            vb.s[3] = *(const short8*)(vp + 24);
            __syncthreads();                      // previous tile's compute done
            *(short8*)(Ks + key * KL + dseg)      = k0;
            *(short8*)(Ks + key * KL + dseg + 8)  = k1;
            *(short8*)(Ks + key * KL + dseg + 16) = k2;
            *(short8*)(Ks + key * KL + dseg + 24) = k3;
#pragma unroll
            for (int j = 0; j < 32; ++j)
                Vs[(dseg + j) * VL + key] = vb.u[j];
            __syncthreads();
        }

        if (kb <= qrow0 + 15) {                   // wave has unmasked keys in this tile
            // S = Q K^T  (4 col-tiles x 4 K-chunks)
            v8f s[4];
#pragma unroll
            for (int nt = 0; nt < 4; ++nt) s[nt] = vzero8();
            FragAB kf;
#pragma unroll
            for (int nt = 0; nt < 4; ++nt) {
#pragma unroll
                for (int c = 0; c < 4; ++c) {
                    const unsigned short* p = Ks + (nt * 16 + lrow) * KL + c * 32 + hi * 8;
                    kf.h[0] = *(const short8*)p;
                    kf.h[1] = *(const short8*)(p + 16);
                    s[nt] = wmma_bf16(qf[c], kf, s[nt]);
                }
            }

            // scale + causal mask + running-max update
            float pv[4][8], tmax[8];
#pragma unroll
            for (int i = 0; i < 8; ++i) tmax[i] = -3.0e38f;
#pragma unroll
            for (int nt = 0; nt < 4; ++nt) {
                const int keyc = kb + nt * 16 + lrow;
#pragma unroll
                for (int i = 0; i < 8; ++i) {
                    const int qr = qrow0 + hi * 8 + i;
                    float v = s[nt][i] * scale;
                    if (keyc > qr) v = -3.0e38f;
                    pv[nt][i] = v;
                    tmax[i] = fmaxf(tmax[i], v);
                }
            }
#pragma unroll
            for (int i = 0; i < 8; ++i) {
                tmax[i] = fmaxf(tmax[i], __shfl_xor(tmax[i], 1, 32));
                tmax[i] = fmaxf(tmax[i], __shfl_xor(tmax[i], 2, 32));
                tmax[i] = fmaxf(tmax[i], __shfl_xor(tmax[i], 4, 32));
                tmax[i] = fmaxf(tmax[i], __shfl_xor(tmax[i], 8, 32));
            }
#pragma unroll
            for (int i = 0; i < 8; ++i) {
                const float mn = fmaxf(rm[i], tmax[i]);
                const float sc = __expf(rm[i] - mn);
                rm[i] = mn;
                rl[i] *= sc;
#pragma unroll
                for (int t = 0; t < 8; ++t) o[t][i] *= sc;
            }
            float psum[8];
#pragma unroll
            for (int i = 0; i < 8; ++i) psum[i] = 0.f;
#pragma unroll
            for (int nt = 0; nt < 4; ++nt)
#pragma unroll
                for (int i = 0; i < 8; ++i) {
                    const float p = __expf(pv[nt][i] - rm[i]);
                    pv[nt][i] = p;
                    psum[i] += p;
                }
#pragma unroll
            for (int i = 0; i < 8; ++i) {
                psum[i] += __shfl_xor(psum[i], 1, 32);
                psum[i] += __shfl_xor(psum[i], 2, 32);
                psum[i] += __shfl_xor(psum[i], 4, 32);
                psum[i] += __shfl_xor(psum[i], 8, 32);
                rl[i] += psum[i];
            }

            // Stage P (C layout -> A layout) in per-wave LDS region
            unsigned short* pw = Ps + w * 16 * VL;
#pragma unroll
            for (int nt = 0; nt < 4; ++nt)
#pragma unroll
                for (int i = 0; i < 8; ++i)
                    pw[(hi * 8 + i) * VL + nt * 16 + lrow] = f2bf(pv[nt][i]);

            FragAB pf[2];
#pragma unroll
            for (int c = 0; c < 2; ++c) {
                const unsigned short* p = pw + lrow * VL + c * 32 + hi * 8;
                pf[c].h[0] = *(const short8*)p;
                pf[c].h[1] = *(const short8*)(p + 16);
            }
            // O += P V  (8 d-tiles x 2 key-chunks)
            FragAB vf;
#pragma unroll
            for (int dt = 0; dt < 8; ++dt) {
#pragma unroll
                for (int c = 0; c < 2; ++c) {
                    const unsigned short* p = Vs + (dt * 16 + lrow) * VL + c * 32 + hi * 8;
                    vf.h[0] = *(const short8*)p;
                    vf.h[1] = *(const short8*)(p + 16);
                    o[dt] = wmma_bf16(pf[c], vf, o[dt]);
                }
            }
        }
    }

#pragma unroll
    for (int i = 0; i < 8; ++i) rl[i] = 1.0f / rl[i];
#pragma unroll
    for (int dt = 0; dt < 8; ++dt)
#pragma unroll
        for (int i = 0; i < 8; ++i) {
            const long r = rowBase + qrow0 + hi * 8 + i;
            outb[r * (long)C + h * D + dt * 16 + lrow] = f2bf(o[dt][i] * rl[i]);
        }
}

// ---------------------------------------------------------------------------
// LayerNorm over rows of C=2048, fp32 out (may alias input) + optional bf16 out
// ---------------------------------------------------------------------------
__global__ __launch_bounds__(256)
void layernorm_fused(const float* __restrict__ x, const float* __restrict__ g,
                     const float* __restrict__ bb, float* __restrict__ yF,
                     unsigned short* __restrict__ yB, int C)
{
    __shared__ float red[256];
    const int tid = threadIdx.x;
    const long row = blockIdx.x;
    const float* xr = x + row * (long)C;
    const float invC = 1.0f / (float)C;
    float lv[8];
    float s = 0.f;
#pragma unroll
    for (int j = 0; j < 8; ++j) { lv[j] = xr[tid + j * 256]; s += lv[j]; }
    red[tid] = s; __syncthreads();
    for (int st = 128; st > 0; st >>= 1) {
        if (tid < st) red[tid] += red[tid + st];
        __syncthreads();
    }
    const float mu = red[0] * invC;
    __syncthreads();
    s = 0.f;
#pragma unroll
    for (int j = 0; j < 8; ++j) { const float d = lv[j] - mu; s += d * d; }
    red[tid] = s; __syncthreads();
    for (int st = 128; st > 0; st >>= 1) {
        if (tid < st) red[tid] += red[tid + st];
        __syncthreads();
    }
    const float rs = rsqrtf(red[0] * invC + 1e-5f);
#pragma unroll
    for (int j = 0; j < 8; ++j) {
        const int c = tid + j * 256;
        const float v = (lv[j] - mu) * rs * g[c] + bb[c];
        if (yF) yF[row * (long)C + c] = v;
        if (yB) yB[row * (long)C + c] = f2bf(v);
    }
}

// ---------------------------------------------------------------------------
extern "C" void kernel_launch(void* const* d_in, const int* in_sizes, int n_in,
                              void* d_out, int out_size, void* d_ws, size_t ws_size,
                              hipStream_t stream) {
    (void)in_sizes; (void)n_in; (void)out_size; (void)ws_size;

    const float* x    = (const float*)d_in[0];
    const float* Wqkv = (const float*)d_in[1];
    const float* bqkv = (const float*)d_in[2];
    const float* Wap  = (const float*)d_in[3];
    const float* bap  = (const float*)d_in[4];
    const float* ln1g = (const float*)d_in[5];
    const float* ln1b = (const float*)d_in[6];
    const float* Wff  = (const float*)d_in[7];
    const float* bff  = (const float*)d_in[8];
    const float* Wfp  = (const float*)d_in[9];
    const float* bfp  = (const float*)d_in[10];
    const float* ln2g = (const float*)d_in[11];
    const float* ln2b = (const float*)d_in[12];
    float* out = (float*)d_out;

    const int B = 4, T = 2048, C = 2048, H3 = 3 * C, FF = 4 * C, NH = 16;
    const int M = B * T;                                   // 8192

    // Workspace bump allocator (aliased buffers; ~336 MB total)
    char* ws = (char*)d_ws;
    size_t off = 0;
    auto alloc = [&](size_t bytes) -> char* {
        char* p = ws + off;
        off += (bytes + 255) & ~(size_t)255;
        return p;
    };
    unsigned short* wqkvT = (unsigned short*)alloc((size_t)C * H3 * 2);
    unsigned short* wapT  = (unsigned short*)alloc((size_t)C * C  * 2);
    unsigned short* wffT  = (unsigned short*)alloc((size_t)C * FF * 2);
    unsigned short* wfpT  = (unsigned short*)alloc((size_t)FF * C * 2);
    unsigned short* xb    = (unsigned short*)alloc((size_t)M * C * 2);    // also x1b
    float*          x1f   = (float*)alloc((size_t)M * C * 4);             // also ln1 out
    char* region          = alloc((size_t)M * FF * 2);                    // qkv+attn, later h
    unsigned short* qkvb  = (unsigned short*)region;
    unsigned short* attnb = (unsigned short*)(region + (size_t)M * H3 * 2);
    unsigned short* hb    = (unsigned short*)region;                      // overlays qkv+attn
    unsigned short* x1b   = xb;                                           // overlays xb

    const dim3 blk(256);

    // Weight conversion + pre-transpose (amortized, bandwidth-trivial vs 1 TFLOP)
    cvt_transpose_kernel<<<dim3(H3 / 32, C / 32), blk, 0, stream>>>(Wqkv, wqkvT, C, H3);
    cvt_transpose_kernel<<<dim3(C / 32,  C / 32), blk, 0, stream>>>(Wap,  wapT,  C, C);
    cvt_transpose_kernel<<<dim3(FF / 32, C / 32), blk, 0, stream>>>(Wff,  wffT,  C, FF);
    cvt_transpose_kernel<<<dim3(C / 32, FF / 32), blk, 0, stream>>>(Wfp,  wfpT,  FF, C);
    cvt_bf16_kernel<<<((long)M * C / 4 + 255) / 256, blk, 0, stream>>>(x, xb, (long)M * C / 4);

    // qkv = x @ W_qkv + b_qkv
    gemm_bf16_wmma<<<dim3(H3 / 128, M / 128), blk, 0, stream>>>(
        xb, wqkvT, bqkv, nullptr, nullptr, qkvb, M, H3, C, 0);

    // causal MHA
    attention_wmma<<<dim3(T / 128, NH, B), blk, 0, stream>>>(qkvb, attnb, T, C, NH);

    // x1 = x + attn @ W_attn_proj + b
    gemm_bf16_wmma<<<dim3(C / 128, M / 128), blk, 0, stream>>>(
        attnb, wapT, bap, x, x1f, nullptr, M, C, C, 0);

    // LN1 (in place, fp32 + bf16)
    layernorm_fused<<<M, blk, 0, stream>>>(x1f, ln1g, ln1b, x1f, x1b, C);

    // h = gelu(x1 @ W_ff + b_ff)
    gemm_bf16_wmma<<<dim3(FF / 128, M / 128), blk, 0, stream>>>(
        x1b, wffT, bff, nullptr, nullptr, hb, M, FF, C, 1);

    // x2 = x1 + h @ W_ff_proj + b  (written straight to d_out)
    gemm_bf16_wmma<<<dim3(C / 128, M / 128), blk, 0, stream>>>(
        hb, wfpT, bfp, x1f, out, nullptr, M, C, FF, 0);

    // LN2 in place on d_out
    layernorm_fused<<<M, blk, 0, stream>>>(out, ln2g, ln2b, out, nullptr, C);
}
